// CrossModalityPositionAttention_56556129354448
// MI455X (gfx1250) — matrix operand order repaired and checked by reading, not compiled, not run on hardware
//
#include <hip/hip_runtime.h>
#include <hip/hip_bf16.h>

// ---------------------------------------------------------------------------
// CrossModalityPositionAttention for MI455X (gfx1250, wave32, WMMA)
// B=4, C=256, c=64, H=W=64, N=4096.
// All GEMMs: v_wmma_f32_16x16x32_bf16 (bf16 in, fp32 accumulate).
// Attention: flash-style (no NxN materialization) with double-buffered
// GLOBAL_LOAD_ASYNC_TO_LDS staging of K/V tiles (ASYNCcnt-tracked),
// issued via inline asm (clang builtin uses unspellable AS pointer types).
// ---------------------------------------------------------------------------

typedef __attribute__((ext_vector_type(16))) __bf16 v16bf;
typedef __attribute__((ext_vector_type(8)))  __bf16 v8bf;
typedef __attribute__((ext_vector_type(8)))  float  v8f;

union ABf { v16bf v; v8bf h2[2]; };   // assemble fragment from 2x16B chunks

__device__ __forceinline__ v8f wmma_bf16(v16bf a, v16bf b, v8f c) {
  // (neg_a, A, neg_b, B, c_mod, C, reuse_a, reuse_b)
  return __builtin_amdgcn_wmma_f32_16x16x32_bf16(false, a, false, b,
                                                 (short)0, c, false, false);
}

// ---- async global->LDS copy (CDNA5: GLOBAL_LOAD_ASYNC_TO_LDS_B128) --------
// VDST = per-lane LDS byte address, VADDR = per-lane 64-bit global address.
__device__ __forceinline__ void async_cp_b128(const void* g, void* l) {
  unsigned lofs =
      (unsigned)(unsigned long long)(__attribute__((address_space(3))) void*)l;
  unsigned long long ga = (unsigned long long)(unsigned long)g;
  asm volatile("global_load_async_to_lds_b128 %0, %1, off"
               :: "v"(lofs), "v"(ga) : "memory");
}

__device__ __forceinline__ void wait_async0() {
  asm volatile("s_wait_asynccnt 0x0" ::: "memory");
}

static constexpr int Bn = 4;     // batch
static constexpr int Cc = 256;   // channels
static constexpr int Cs = 64;    // c = C/4
static constexpr int Hh = 64, Ww = 64;
static constexpr int Nn = Hh * Ww;          // 4096
static constexpr float EPS = 1e-5f;

// ---------------------------------------------------------------------------
// Kernel 0: NCHW fp32 -> NHWC bf16 (LDS tiled transpose). One block per (b,h).
// ---------------------------------------------------------------------------
__global__ __launch_bounds__(256)
void nchw_to_nhwc_bf16_exact(const float* __restrict__ in, __bf16* __restrict__ out) {
  __shared__ float tile[32][65];
  const int b = blockIdx.x >> 6;
  const int h = blockIdx.x & 63;
  const int t = threadIdx.x;
  for (int ci0 = 0; ci0 < Cc; ci0 += 32) {
#pragma unroll
    for (int r = 0; r < 8; ++r) {           // read 32ci x 64w, coalesced in w
      int idx = r * 256 + t;
      int ci = idx >> 6, w = idx & 63;
      tile[ci][w] = in[(((size_t)b * Cc + ci0 + ci) * Hh + h) * Ww + w];
    }
    __syncthreads();
#pragma unroll
    for (int r = 0; r < 8; ++r) {           // write coalesced in ci
      int idx = r * 256 + t;
      int ci = idx & 31, w = idx >> 5;
      out[(((size_t)b * Hh + h) * Ww + w) * Cc + ci0 + ci] = (__bf16)tile[ci][w];
    }
    __syncthreads();
  }
}

// ---------------------------------------------------------------------------
// Kernel 1: weight repack OIHW fp32 -> [O][kh][kw][I] bf16 (A-operand rows
// become K-contiguous).
// ---------------------------------------------------------------------------
__global__ __launch_bounds__(256)
void repack_oihw(const float* __restrict__ in, __bf16* __restrict__ out,
                 int O, int I) {
  int total = O * I * 9;
  for (int idx = blockIdx.x * 256 + threadIdx.x; idx < total;
       idx += gridDim.x * 256) {
    int i  = idx % I;
    int r  = idx / I;
    int kw = r % 3; r /= 3;
    int kh = r % 3;
    int o  = r / 3;
    out[idx] = (__bf16)in[(((size_t)o * I + i) * 3 + kh) * 3 + kw];
  }
}

// ---------------------------------------------------------------------------
// Kernel 2: conv3x3(256->64) + BN + ReLU, implicit GEMM via WMMA.
// Block = 256 thr = 8 waves; wave -> (co tile of 16) x (w tile of 16).
// Output layout selectable by strides: q/k -> [b][n][c], v -> [b][c][n].
// ---------------------------------------------------------------------------
__global__ __launch_bounds__(256)
void conv_qkv_wmma(const __bf16* __restrict__ featN,   // [B][H][W][256]
                   const __bf16* __restrict__ wp,      // [64][3][3][256]
                   const float* __restrict__ bias, const float* __restrict__ gamma,
                   const float* __restrict__ beta, const float* __restrict__ mean,
                   const float* __restrict__ var,
                   __bf16* __restrict__ outp, int sN, int sC) {
  const int lane = threadIdx.x & 31, wave = threadIdx.x >> 5;
  const int lo = lane & 15, hi = lane >> 4;
  const int b = blockIdx.z, h = blockIdx.y;
  const int w0  = blockIdx.x * 32 + (wave >> 2) * 16;
  const int co0 = (wave & 3) * 16;

  v8f acc = (v8f)0.0f;
  const __bf16* wbase = wp + ((size_t)(co0 + lo) * 9) * Cc + hi * 16;

  for (int dh = 0; dh < 3; ++dh) {
    int hin = h + dh - 1;
    for (int dw = 0; dw < 3; ++dw) {
      int win = w0 + lo + dw - 1;
      bool inb = (hin >= 0) && (hin < Hh) && (win >= 0) && (win < Ww);
      int hc = inb ? hin : 0, wc = inb ? win : 0;
      const __bf16* frow = featN + (((size_t)b * Hh + hc) * Ww + wc) * Cc + hi * 16;
      const __bf16* wrow = wbase + (size_t)(dh * 3 + dw) * Cc;
#pragma unroll
      for (int ci0 = 0; ci0 < Cc; ci0 += 32) {
        ABf A;                                   // A: 16 co rows x 32 ci
        A.h2[0] = *(const v8bf*)(wrow + ci0);
        A.h2[1] = *(const v8bf*)(wrow + ci0 + 8);
        v16bf Bv = (v16bf)(__bf16)0.0f;          // B: 32 ci x 16 spatial
        if (inb) Bv = *(const v16bf*)(frow + ci0);
        acc = wmma_bf16(A.v, Bv, acc);
      }
    }
  }
  const int n = h * Ww + w0 + lo;
#pragma unroll
  for (int r = 0; r < 8; ++r) {
    int co = co0 + r + 8 * hi;
    float inv = gamma[co] * rsqrtf(var[co] + EPS);
    float val = (acc[r] + bias[co]) * inv + (beta[co] - mean[co] * inv);
    val = fmaxf(val, 0.0f);
    outp[(size_t)b * (Nn * Cs) + (size_t)n * sN + (size_t)co * sC] = (__bf16)val;
  }
}

// ---------------------------------------------------------------------------
// Kernel 3: flash attention with async-LDS K/V staging (double-buffered).
//   kbuf rows: 72 halves (144B stride, 16B-aligned chunks, bank-spread)
//   vbuf rows: 56 halves (112B stride)
// ---------------------------------------------------------------------------
__global__ __launch_bounds__(256)
void flash_attn_wmma(const __bf16* __restrict__ qT,  // [B][N][64]
                     const __bf16* __restrict__ kT,  // [B][N][64]
                     const __bf16* __restrict__ vC,  // [B][64][N]
                     __bf16* __restrict__ fT) {      // [B][N][64]
  __shared__ __bf16 kbuf[2][32][72];
  __shared__ __bf16 vbuf[2][64][56];
  __shared__ __bf16 psc[8][16 * 48];
  const int t = threadIdx.x;
  const int lane = t & 31, wave = t >> 5;
  const int lo = lane & 15, hi = lane >> 4;
  const int b  = blockIdx.y;
  const int n0 = blockIdx.x * 128 + wave * 16;

  const __bf16* kb = kT + (size_t)b * Nn * Cs;
  const __bf16* vb = vC + (size_t)b * Cs * Nn;

  // Persistent A operand: 16 query rows x 64 c (two 16x32 k-steps)
  ABf aq0, aq1;
  {
    const __bf16* qrow = qT + ((size_t)b * Nn + n0 + lo) * Cs + hi * 16;
    aq0.h2[0] = *(const v8bf*)(qrow);      aq0.h2[1] = *(const v8bf*)(qrow + 8);
    aq1.h2[0] = *(const v8bf*)(qrow + 32); aq1.h2[1] = *(const v8bf*)(qrow + 40);
  }

  // async tile copy: 256 threads x (1 K-chunk + 1 V-chunk) x 16B
  const int kr = t >> 3, kc = (t & 7) * 8;   // K: 32 rows x 64 c
  const int vr = t >> 2, vc = (t & 3) * 8;   // V: 64 rows x 32 m
  auto issue_tile = [&](int buf, int m0) {
    async_cp_b128(kb + (size_t)(m0 + kr) * Cs + kc, &kbuf[buf][kr][kc]);
    async_cp_b128(vb + (size_t)vr * Nn + m0 + vc,   &vbuf[buf][vr][vc]);
  };

  float rm[8], rs[8];
  v8f O[4];
#pragma unroll
  for (int r = 0; r < 8; ++r) { rm[r] = -3.0e38f; rs[r] = 0.0f; }
#pragma unroll
  for (int ct = 0; ct < 4; ++ct) O[ct] = (v8f)0.0f;

  __bf16* ps = psc[wave];

  issue_tile(0, 0);                       // prologue

  for (int it = 0; it < Nn / 32; ++it) {
    const int buf = it & 1;
    // tile[it] resident + every wave done reading buf^1 (tile it-1):
    wait_async0();
    __syncthreads();
    if (it + 1 < Nn / 32) issue_tile(buf ^ 1, (it + 1) * 32);  // overlap copy

    // ---- S = Q^T K for 16n x 32m (B fragments from LDS) ----
    ABf bk00, bk01, bk10, bk11;
    bk00.h2[0] = *(const v8bf*)(&kbuf[buf][lo][hi * 16]);
    bk00.h2[1] = *(const v8bf*)(&kbuf[buf][lo][hi * 16 + 8]);
    bk01.h2[0] = *(const v8bf*)(&kbuf[buf][lo][32 + hi * 16]);
    bk01.h2[1] = *(const v8bf*)(&kbuf[buf][lo][32 + hi * 16 + 8]);
    bk10.h2[0] = *(const v8bf*)(&kbuf[buf][16 + lo][hi * 16]);
    bk10.h2[1] = *(const v8bf*)(&kbuf[buf][16 + lo][hi * 16 + 8]);
    bk11.h2[0] = *(const v8bf*)(&kbuf[buf][16 + lo][32 + hi * 16]);
    bk11.h2[1] = *(const v8bf*)(&kbuf[buf][16 + lo][32 + hi * 16 + 8]);
    v8f s1 = (v8f)0.0f, s2 = (v8f)0.0f;
    s1 = wmma_bf16(aq0.v, bk00.v, s1); s1 = wmma_bf16(aq1.v, bk01.v, s1);
    s2 = wmma_bf16(aq0.v, bk10.v, s2); s2 = wmma_bf16(aq1.v, bk11.v, s2);

    // ---- online softmax (row r+8*hi lives across the lane's 16-group) ----
    float sc[8];
#pragma unroll
    for (int r = 0; r < 8; ++r) {
      float tmax = fmaxf(s1[r], s2[r]);
#pragma unroll
      for (int m = 1; m < 16; m <<= 1) tmax = fmaxf(tmax, __shfl_xor(tmax, m, 32));
      float nm = fmaxf(rm[r], tmax);
      sc[r] = __expf(rm[r] - nm);
      rm[r] = nm;
    }
#pragma unroll
    for (int r = 0; r < 8; ++r) {
      s1[r] = __expf(s1[r] - rm[r]);
      s2[r] = __expf(s2[r] - rm[r]);
      float tsum = s1[r] + s2[r];
#pragma unroll
      for (int m = 1; m < 16; m <<= 1) tsum += __shfl_xor(tsum, m, 32);
      rs[r] = rs[r] * sc[r] + tsum;
    }
#pragma unroll
    for (int ct = 0; ct < 4; ++ct)
#pragma unroll
      for (int r = 0; r < 8; ++r) O[ct][r] *= sc[r];

    // ---- transpose P (C/D layout -> A layout) via per-wave LDS scratch ----
#pragma unroll
    for (int r = 0; r < 8; ++r) {
      int row = r + 8 * hi;
      ps[row * 48 + lo]      = (__bf16)s1[r];
      ps[row * 48 + 16 + lo] = (__bf16)s2[r];
    }
    asm volatile("s_wait_dscnt 0x0" ::: "memory");
    ABf ap;
    const __bf16* prow = ps + lo * 48 + hi * 16;
    ap.h2[0] = *(const v8bf*)(prow);
    ap.h2[1] = *(const v8bf*)(prow + 8);

    // ---- O += P * V^T (K-dim = m, contiguous per lane in LDS [c][m]) ----
#pragma unroll
    for (int ct = 0; ct < 4; ++ct) {
      ABf bv;
      bv.h2[0] = *(const v8bf*)(&vbuf[buf][ct * 16 + lo][hi * 16]);
      bv.h2[1] = *(const v8bf*)(&vbuf[buf][ct * 16 + lo][hi * 16 + 8]);
      O[ct] = wmma_bf16(ap.v, bv.v, O[ct]);
    }
  }

#pragma unroll
  for (int r = 0; r < 8; ++r) rs[r] = 1.0f / rs[r];
#pragma unroll
  for (int ct = 0; ct < 4; ++ct)
#pragma unroll
    for (int r = 0; r < 8; ++r) {
      int n = n0 + r + 8 * hi;
      fT[((size_t)b * Nn + n) * Cs + ct * 16 + lo] = (__bf16)(O[ct][r] * rs[r]);
    }
}

// ---------------------------------------------------------------------------
// Kernel 4: conv3x3(64->256) + BN + ReLU + residual, fp32 NCHW output.
// Wave handles two co tiles (co, co+128) sharing the B operand.
// ---------------------------------------------------------------------------
__global__ __launch_bounds__(256)
void conv_r_wmma(const __bf16* __restrict__ fT,   // [B][N][64] (NHWC-like)
                 const __bf16* __restrict__ wp,   // [256][3][3][64]
                 const float* __restrict__ bias, const float* __restrict__ gamma,
                 const float* __restrict__ beta, const float* __restrict__ mean,
                 const float* __restrict__ var,
                 const float* __restrict__ feat1, // [B][256][H][W]
                 float* __restrict__ out) {
  const int lane = threadIdx.x & 31, wave = threadIdx.x >> 5;
  const int lo = lane & 15, hi = lane >> 4;
  const int b = blockIdx.z, h = blockIdx.y;
  const int w0 = blockIdx.x * 16;
  const int coA = wave * 16, coB = coA + 128;

  v8f acc0 = (v8f)0.0f, acc1 = (v8f)0.0f;
  const __bf16* wbA = wp + ((size_t)(coA + lo) * 9) * Cs + hi * 16;
  const __bf16* wbB = wp + ((size_t)(coB + lo) * 9) * Cs + hi * 16;

  for (int dh = 0; dh < 3; ++dh) {
    int hin = h + dh - 1;
    for (int dw = 0; dw < 3; ++dw) {
      int win = w0 + lo + dw - 1;
      bool inb = (hin >= 0) && (hin < Hh) && (win >= 0) && (win < Ww);
      int hc = inb ? hin : 0, wc = inb ? win : 0;
      const __bf16* frow = fT + ((size_t)b * Nn + hc * Ww + wc) * Cs + hi * 16;
      const __bf16* wrA = wbA + (size_t)(dh * 3 + dw) * Cs;
      const __bf16* wrB = wbB + (size_t)(dh * 3 + dw) * Cs;
#pragma unroll
      for (int ci0 = 0; ci0 < Cs; ci0 += 32) {
        v16bf Bv = (v16bf)(__bf16)0.0f;
        if (inb) Bv = *(const v16bf*)(frow + ci0);
        ABf A0, A1;
        A0.h2[0] = *(const v8bf*)(wrA + ci0); A0.h2[1] = *(const v8bf*)(wrA + ci0 + 8);
        A1.h2[0] = *(const v8bf*)(wrB + ci0); A1.h2[1] = *(const v8bf*)(wrB + ci0 + 8);
        acc0 = wmma_bf16(A0.v, Bv, acc0);
        acc1 = wmma_bf16(A1.v, Bv, acc1);
      }
    }
  }
  const int w = w0 + lo;
#pragma unroll
  for (int r = 0; r < 8; ++r) {
    int co = coA + r + 8 * hi;
    float inv = gamma[co] * rsqrtf(var[co] + EPS);
    float val = (acc0[r] + bias[co]) * inv + (beta[co] - mean[co] * inv);
    val = fmaxf(val, 0.0f);
    size_t o = (((size_t)b * Cc + co) * Hh + h) * Ww + w;
    out[o] = feat1[o] + val;
  }
#pragma unroll
  for (int r = 0; r < 8; ++r) {
    int co = coB + r + 8 * hi;
    float inv = gamma[co] * rsqrtf(var[co] + EPS);
    float val = (acc1[r] + bias[co]) * inv + (beta[co] - mean[co] * inv);
    val = fmaxf(val, 0.0f);
    size_t o = (((size_t)b * Cc + co) * Hh + h) * Ww + w;
    out[o] = feat1[o] + val;
  }
}

// ---------------------------------------------------------------------------
// Host launcher
// ---------------------------------------------------------------------------
extern "C" void kernel_launch(void* const* d_in, const int* in_sizes, int n_in,
                              void* d_out, int out_size, void* d_ws, size_t ws_size,
                              hipStream_t stream) {
  const float* feature1 = (const float*)d_in[0];
  const float* feature2 = (const float*)d_in[1];
  const float *qw = (const float*)d_in[2],  *qb = (const float*)d_in[3],
              *qg = (const float*)d_in[4],  *qbe = (const float*)d_in[5],
              *qm = (const float*)d_in[6],  *qv = (const float*)d_in[7];
  const float *kw_ = (const float*)d_in[8], *kb = (const float*)d_in[9],
              *kg = (const float*)d_in[10], *kbe = (const float*)d_in[11],
              *km = (const float*)d_in[12], *kv = (const float*)d_in[13];
  const float *vw = (const float*)d_in[14], *vb = (const float*)d_in[15],
              *vg = (const float*)d_in[16], *vbe = (const float*)d_in[17],
              *vm = (const float*)d_in[18], *vv = (const float*)d_in[19];
  const float *rw = (const float*)d_in[20], *rb = (const float*)d_in[21],
              *rg = (const float*)d_in[22], *rbe = (const float*)d_in[23],
              *rm = (const float*)d_in[24], *rv = (const float*)d_in[25];
  float* out = (float*)d_out;

  auto al = [](size_t x) { return (x + 511) & ~(size_t)511; };
  char* base = (char*)d_ws;
  size_t off = 0;
  __bf16* f1n = (__bf16*)(base + off); off = al(off + (size_t)Bn * Nn * Cc * 2);
  __bf16* f2n = (__bf16*)(base + off); off = al(off + (size_t)Bn * Nn * Cc * 2);
  __bf16* wqp = (__bf16*)(base + off); off = al(off + (size_t)Cs * 9 * Cc * 2);
  __bf16* wkp = (__bf16*)(base + off); off = al(off + (size_t)Cs * 9 * Cc * 2);
  __bf16* wvp = (__bf16*)(base + off); off = al(off + (size_t)Cs * 9 * Cc * 2);
  __bf16* wrp = (__bf16*)(base + off); off = al(off + (size_t)Cc * 9 * Cs * 2);
  __bf16* qT  = (__bf16*)(base + off); off = al(off + (size_t)Bn * Nn * Cs * 2);
  __bf16* kT  = (__bf16*)(base + off); off = al(off + (size_t)Bn * Nn * Cs * 2);
  __bf16* vCm = (__bf16*)(base + off); off = al(off + (size_t)Bn * Cs * Nn * 2);
  __bf16* fT  = (__bf16*)(base + off); off = al(off + (size_t)Bn * Nn * Cs * 2);

  // 0) layout transforms
  nchw_to_nhwc_bf16_exact<<<dim3(Bn * Hh), 256, 0, stream>>>(feature1, f1n);
  nchw_to_nhwc_bf16_exact<<<dim3(Bn * Hh), 256, 0, stream>>>(feature2, f2n);
  repack_oihw<<<dim3(576), 256, 0, stream>>>(qw,  wqp, Cs, Cc);
  repack_oihw<<<dim3(576), 256, 0, stream>>>(kw_, wkp, Cs, Cc);
  repack_oihw<<<dim3(576), 256, 0, stream>>>(vw,  wvp, Cs, Cc);
  repack_oihw<<<dim3(576), 256, 0, stream>>>(rw,  wrp, Cc, Cs);

  // 1) q/k/v convs (implicit GEMM, WMMA). q,k -> [n][c]; v -> [c][n].
  dim3 cg(Ww / 32, Hh, Bn);
  conv_qkv_wmma<<<cg, 256, 0, stream>>>(f2n, wqp, qb, qg, qbe, qm, qv, qT, Cs, 1);
  conv_qkv_wmma<<<cg, 256, 0, stream>>>(f1n, wkp, kb, kg, kbe, km, kv, kT, Cs, 1);
  conv_qkv_wmma<<<cg, 256, 0, stream>>>(f1n, wvp, vb, vg, vbe, vm, vv, vCm, 1, Nn);

  // 2) flash attention (128 rows per block, 8 waves x 16 rows, async-LDS K/V)
  flash_attn_wmma<<<dim3(Nn / 128, Bn), 256, 0, stream>>>(qT, kT, vCm, fT);

  // 3) reverse conv + BN + ReLU + residual
  conv_r_wmma<<<dim3(Ww / 16, Hh, Bn), 256, 0, stream>>>(
      fT, wrp, rb, rg, rbe, rm, rv, feature1, out);
}
